// PyTorchVisionAttention_69054484185758
// MI455X (gfx1250) — compile-verified, matrix operand order checked by reading, not run on hardware
//
#include <hip/hip_runtime.h>
#include <hip/hip_bf16.h>

typedef __bf16 bf16;
typedef __attribute__((ext_vector_type(16))) __bf16 v16bf;
typedef __attribute__((ext_vector_type(8)))  __bf16 v8bf;
typedef __attribute__((ext_vector_type(8)))  float  v8f;
typedef __attribute__((ext_vector_type(4)))  int    v4i;

#define S_LEN 2048
#define DMODEL 1280
#define NHEAD 16
#define HDIM 80
#define HPAD 96
#define QKVN 3840
#define KSTEP 32
#define PADROW 40   /* 32 k-elements + 8 pad -> 80B row stride, conflict-free */

// ---- gfx1250 async global->LDS path (ASYNCcnt), guarded for portability ----
#if defined(__has_builtin)
#if __has_builtin(__builtin_amdgcn_global_load_async_to_lds_b128) && \
    __has_builtin(__builtin_amdgcn_s_wait_asynccnt)
#define USE_ASYNC_LDS 1
#endif
#endif
#ifndef USE_ASYNC_LDS
#define USE_ASYNC_LDS 0
#endif

#if USE_ASYNC_LDS
typedef __attribute__((address_space(1))) v4i* as1_v4i_p;
typedef __attribute__((address_space(3))) v4i* as3_v4i_p;
#define CP16(g, l)                                                        \
    __builtin_amdgcn_global_load_async_to_lds_b128(                       \
        (as1_v4i_p)(g), (as3_v4i_p)(l), 0, 0)
#define WAIT_ASYNC(n) __builtin_amdgcn_s_wait_asynccnt(n)
#else
#define CP16(g, l) (*(v8bf*)(l) = *(const v8bf*)(g))
#define WAIT_ASYNC(n) ((void)0)
#endif

__device__ __forceinline__ v8f wmma_bf16(v16bf a, v16bf b, v8f c) {
    return __builtin_amdgcn_wmma_f32_16x16x32_bf16(
        false, a, false, b, (short)0, c, false, false);
}

__device__ __forceinline__ v16bf lds_frag(const bf16* p) {
    union { v16bf v; v8bf h[2]; } u;
    u.h[0] = *(const v8bf*)(p);
    u.h[1] = *(const v8bf*)(p + 8);
    return u.v;
}

// ---------------------------------------------------------------------------
// fp32 -> bf16 conversion
// ---------------------------------------------------------------------------
__global__ void cvt_f32_bf16(const float* __restrict__ src,
                             bf16* __restrict__ dst, int n) {
    int i = blockIdx.x * blockDim.x + threadIdx.x;
    if (i < n) dst[i] = (bf16)src[i];
}

// ---------------------------------------------------------------------------
// Generic TN GEMM: C[m,n] = sum_k A[m,k]*B[n,k] + bias[n]
// A: M x K bf16 row-major, B: N x K bf16 row-major, C: M x N fp32.
// Block = 256 threads (8 waves), tile 128x128, wave tile 64x32.
// K-step 32, double-buffered LDS staged via async global->LDS copies.
// ---------------------------------------------------------------------------
__global__ __launch_bounds__(256) void gemm_tn_bf16(
    const bf16* __restrict__ A, const bf16* __restrict__ B,
    const float* __restrict__ bias, float* __restrict__ C,
    int M, int N, int K)
{
    __shared__ __align__(128) bf16 As[2][128][PADROW];
    __shared__ __align__(128) bf16 Bs[2][128][PADROW];

    const int lane = threadIdx.x & 31;
    const int wave = threadIdx.x >> 5;
    const int nBlk = N >> 7;
    const int mBlk = (blockIdx.x / nBlk) * 128;
    const int nBlk0 = (blockIdx.x % nBlk) * 128;
    const int mW = (wave >> 2) * 64;
    const int nW = (wave & 3) * 32;
    const int lr = lane & 15;
    const int lh = (lane >> 4) * 16;

    // cooperative tile copy: thread t stages row t>>1, 32B chunk t&1
    const int crow = threadIdx.x >> 1;
    const int ccol = (threadIdx.x & 1) * 16;
    const bf16* gA = A + (size_t)(mBlk + crow) * K + ccol;
    const bf16* gB = B + (size_t)(nBlk0 + crow) * K + ccol;

    v8f acc[4][2];
#pragma unroll
    for (int i = 0; i < 4; ++i)
#pragma unroll
        for (int j = 0; j < 2; ++j)
            acc[i][j] = {};

    const int nk = K / KSTEP;

    // prologue: stage k-step 0 into buffer 0
    CP16(gA,     &As[0][crow][ccol]);
    CP16(gA + 8, &As[0][crow][ccol + 8]);
    CP16(gB,     &Bs[0][crow][ccol]);
    CP16(gB + 8, &Bs[0][crow][ccol + 8]);

    for (int ik = 0; ik < nk; ++ik) {
        const int cur = ik & 1;
        const int nxt = cur ^ 1;
        if (ik + 1 < nk) {
            const int k0 = (ik + 1) * KSTEP;
            CP16(gA + k0,     &As[nxt][crow][ccol]);
            CP16(gA + k0 + 8, &As[nxt][crow][ccol + 8]);
            CP16(gB + k0,     &Bs[nxt][crow][ccol]);
            CP16(gB + k0 + 8, &Bs[nxt][crow][ccol + 8]);
            WAIT_ASYNC(4);   // current buffer's 4 copies complete
        } else {
            WAIT_ASYNC(0);
        }
        __syncthreads();     // all waves' copies for `cur` visible

        v16bf af[4], bfr[2];
#pragma unroll
        for (int i = 0; i < 4; ++i)
            af[i] = lds_frag(&As[cur][mW + i * 16 + lr][lh]);
#pragma unroll
        for (int j = 0; j < 2; ++j)
            bfr[j] = lds_frag(&Bs[cur][nW + j * 16 + lr][lh]);
#pragma unroll
        for (int i = 0; i < 4; ++i)
#pragma unroll
            for (int j = 0; j < 2; ++j)
                acc[i][j] = wmma_bf16(af[i], bfr[j], acc[i][j]);

        __syncthreads();     // reads done before `cur` is overwritten next
    }

    const int rbase = (lane >> 4) * 8;
#pragma unroll
    for (int i = 0; i < 4; ++i)
#pragma unroll
        for (int j = 0; j < 2; ++j) {
            const int col = nBlk0 + nW + j * 16 + lr;
            const float bv = bias ? bias[col] : 0.0f;
#pragma unroll
            for (int r = 0; r < 8; ++r) {
                const int row = mBlk + mW + i * 16 + rbase + r;
                C[(size_t)row * N + col] = acc[i][j][r] + bv;
            }
        }
}

// ---------------------------------------------------------------------------
// RoPE + layout transform.
// qkv_ws: [S][3840] fp32  ->  Qh,Kh: [H][S][HPAD] bf16 (rope applied,
// dims 80..95 zero),  Vt: [H][HDIM][S] bf16 (transposed for P@V B-operand).
// ---------------------------------------------------------------------------
__global__ void rope_pack(const float* __restrict__ qkv,
                          const float* __restrict__ rope,
                          bf16* __restrict__ Qh, bf16* __restrict__ Kh,
                          bf16* __restrict__ Vt)
{
    int idx = blockIdx.x * blockDim.x + threadIdx.x;
    if (idx >= NHEAD * S_LEN * HPAD) return;
    const int dp = idx % HPAD;
    const int s  = (idx / HPAD) % S_LEN;
    const int h  = idx / (HPAD * S_LEN);

    const size_t qkidx = ((size_t)h * S_LEN + s) * HPAD + dp;
    if (dp < HDIM) {
        const int d2 = (dp < 40) ? dp : dp - 40;
        const float ang = rope[s * 40 + d2];
        const float c = cosf(ang), sn = sinf(ang);
        const size_t base = (size_t)s * QKVN + h * HDIM;
        const float qv = qkv[base + dp];
        const float kv = qkv[base + DMODEL + dp];
        const float vv = qkv[base + 2 * DMODEL + dp];
        const float qr = (dp < 40) ? -qkv[base + dp + 40]
                                   :  qkv[base + dp - 40];
        const float kr = (dp < 40) ? -qkv[base + DMODEL + dp + 40]
                                   :  qkv[base + DMODEL + dp - 40];
        Qh[qkidx] = (bf16)(qv * c + qr * sn);
        Kh[qkidx] = (bf16)(kv * c + kr * sn);
        Vt[((size_t)h * HDIM + dp) * S_LEN + s] = (bf16)vv;
    } else {
        Qh[qkidx] = (bf16)0.0f;
        Kh[qkidx] = (bf16)0.0f;
    }
}

// ---------------------------------------------------------------------------
// Flash attention: one wave handles 16 query rows of one head.
// Block = 128 threads (4 waves) -> 64 query rows. Grid = H * (S/64).
// ---------------------------------------------------------------------------
__global__ __launch_bounds__(128) void attn_fwd(
    const bf16* __restrict__ Qh, const bf16* __restrict__ Kh,
    const bf16* __restrict__ Vt, bf16* __restrict__ Ah)
{
    const int lane = threadIdx.x & 31;
    const int wave = threadIdx.x >> 5;
    const int h  = blockIdx.x >> 5;
    const int qb = blockIdx.x & 31;
    const int q0 = qb * 64 + wave * 16;
    const int lr = lane & 15;
    const int lh = lane >> 4;

    __shared__ __align__(64) bf16 pbuf[4][16][32];

    const float scaling = 0.11180339887498948f;   // 80^-0.5

    v16bf qf[3];
    const bf16* qrow = Qh + ((size_t)h * S_LEN + q0 + lr) * HPAD;
#pragma unroll
    for (int kk = 0; kk < 3; ++kk)
        qf[kk] = *(const v16bf*)(qrow + kk * 32 + lh * 16);

    float mrow[8], lrow[8];
    v8f acc[5];
#pragma unroll
    for (int r = 0; r < 8; ++r) { mrow[r] = -1e30f; lrow[r] = 0.0f; }
#pragma unroll
    for (int t = 0; t < 5; ++t) acc[t] = {};

    for (int kb = 0; kb < S_LEN / 32; ++kb) {
        const int key0 = kb * 32;

        v8f s[2];
#pragma unroll
        for (int c = 0; c < 2; ++c) {
            const bf16* krow = Kh + ((size_t)h * S_LEN + key0 + c*16 + lr) * HPAD;
            v8f t = {};
#pragma unroll
            for (int kk = 0; kk < 3; ++kk) {
                v16bf kf = *(const v16bf*)(krow + kk * 32 + lh * 16);
                t = wmma_bf16(qf[kk], kf, t);
            }
            s[c] = t;
        }

#pragma unroll
        for (int r = 0; r < 8; ++r) {
            float x0 = s[0][r] * scaling;
            float x1 = s[1][r] * scaling;
            float mx = fmaxf(x0, x1);
#pragma unroll
            for (int d = 8; d >= 1; d >>= 1)
                mx = fmaxf(mx, __shfl_xor(mx, d, 32));
            const float mnew = fmaxf(mrow[r], mx);
            const float alpha = __expf(mrow[r] - mnew);
            const float p0 = __expf(x0 - mnew);
            const float p1 = __expf(x1 - mnew);
            float rs = p0 + p1;
#pragma unroll
            for (int d = 8; d >= 1; d >>= 1)
                rs += __shfl_xor(rs, d, 32);
            lrow[r] = lrow[r] * alpha + rs;
            mrow[r] = mnew;
#pragma unroll
            for (int t = 0; t < 5; ++t) acc[t][r] *= alpha;

            const int prow = r + 8 * lh;
            pbuf[wave][prow][lr]      = (bf16)p0;
            pbuf[wave][prow][16 + lr] = (bf16)p1;
        }

        v16bf pf = lds_frag(&pbuf[wave][lr][lh * 16]);
#pragma unroll
        for (int t = 0; t < 5; ++t) {
            const bf16* vrow = Vt + ((size_t)h * HDIM + t*16 + lr) * S_LEN
                               + key0 + lh * 16;
            v16bf vf = *(const v16bf*)vrow;
            acc[t] = wmma_bf16(pf, vf, acc[t]);
        }
    }

#pragma unroll
    for (int t = 0; t < 5; ++t)
#pragma unroll
        for (int r = 0; r < 8; ++r) {
            const float o = acc[t][r] / lrow[r];
            const int row = q0 + r + 8 * lh;
            const int col = h * HDIM + t * 16 + lr;
            Ah[(size_t)row * DMODEL + col] = (bf16)o;
        }
}

// ---------------------------------------------------------------------------
// Host launch
// ---------------------------------------------------------------------------
extern "C" void kernel_launch(void* const* d_in, const int* in_sizes, int n_in,
                              void* d_out, int out_size, void* d_ws, size_t ws_size,
                              hipStream_t stream) {
    const float* hidden = (const float*)d_in[0];
    const float* rope   = (const float*)d_in[1];
    const float* qkv_w  = (const float*)d_in[2];
    const float* qkv_b  = (const float*)d_in[3];
    const float* proj_w = (const float*)d_in[4];
    const float* proj_b = (const float*)d_in[5];
    float* out = (float*)d_out;

    char* ws = (char*)d_ws;
    size_t off = 0;
    auto alloc = [&](size_t bytes) { char* p = ws + off; off += (bytes + 255) & ~(size_t)255; return p; };

    bf16*  hidden_bf = (bf16*)alloc((size_t)S_LEN * DMODEL * 2);
    bf16*  qkvw_bf   = (bf16*)alloc((size_t)QKVN * DMODEL * 2);
    bf16*  projw_bf  = (bf16*)alloc((size_t)DMODEL * DMODEL * 2);
    float* qkv_ws    = (float*)alloc((size_t)S_LEN * QKVN * 4);
    bf16*  Qh        = (bf16*)alloc((size_t)NHEAD * S_LEN * HPAD * 2);
    bf16*  Kh        = (bf16*)alloc((size_t)NHEAD * S_LEN * HPAD * 2);
    bf16*  Vt        = (bf16*)alloc((size_t)NHEAD * HDIM * S_LEN * 2);
    bf16*  Ah        = (bf16*)alloc((size_t)S_LEN * DMODEL * 2);
    (void)ws_size; (void)n_in; (void)in_sizes; (void)out_size;

    {
        int n1 = S_LEN * DMODEL;
        cvt_f32_bf16<<<(n1 + 255) / 256, 256, 0, stream>>>(hidden, hidden_bf, n1);
        int n2 = QKVN * DMODEL;
        cvt_f32_bf16<<<(n2 + 255) / 256, 256, 0, stream>>>(qkv_w, qkvw_bf, n2);
        int n3 = DMODEL * DMODEL;
        cvt_f32_bf16<<<(n3 + 255) / 256, 256, 0, stream>>>(proj_w, projw_bf, n3);
    }

    {
        dim3 grid((S_LEN / 128) * (QKVN / 128));
        gemm_tn_bf16<<<grid, 256, 0, stream>>>(hidden_bf, qkvw_bf, qkv_b, qkv_ws,
                                               S_LEN, QKVN, DMODEL);
    }

    {
        int n = NHEAD * S_LEN * HPAD;
        rope_pack<<<(n + 255) / 256, 256, 0, stream>>>(qkv_ws, rope, Qh, Kh, Vt);
    }

    {
        dim3 grid(NHEAD * (S_LEN / 64));
        attn_fwd<<<grid, 128, 0, stream>>>(Qh, Kh, Vt, Ah);
    }

    {
        dim3 grid((S_LEN / 128) * (DMODEL / 128));
        gemm_tn_bf16<<<grid, 256, 0, stream>>>(Ah, projw_bf, proj_b, out,
                                               S_LEN, DMODEL, DMODEL);
    }
}
